// RFCNRoIhead_24300924961347
// MI455X (gfx1250) — matrix operand order repaired and checked by reading, not compiled, not run on hardware
//
#include <hip/hip_runtime.h>
#include <hip/hip_bf16.h>

// ---------------------------------------------------------------------------
// RFCN RoI head, restructured:
//   1) SAT (summed-area table) of h            [2,65,65,1024] f32
//   2) PSRoI bin-averages per roi/channel      P [1024, 49*1024] bf16
//   3) fold 1/49 into weights                  Bmat [112, 50176] bf16 (K-major)
//   4) GEMM 1024x50176x112 via v_wmma_f32_16x16x32_bf16 (K split 16-way)
//   5) deterministic K-split reduction + bias + scatter to output layout
// ---------------------------------------------------------------------------

typedef __attribute__((ext_vector_type(16))) __bf16 v16bf;
typedef __attribute__((ext_vector_type(8)))  __bf16 v8bf;
typedef __attribute__((ext_vector_type(8)))  float  v8f;

#define NB    2
#define CIN   1024
#define HH    64
#define WW    64
#define RR    1024
#define BINSK 7
#define NBIN  49            // 7*7
#define KDIM  50176         // 49*1024
#define NOUT  105           // 21 + 84
#define NPAD  112
#define KSPLIT 16
#define KSEG  3136          // KDIM / KSPLIT = 98 steps of 32
#define SCALE_F 0.0625f

// SAT index: [n][yy][xx][c], c innermost
__device__ __forceinline__ int sat_idx(int n, int yy, int xx, int c) {
    return ((n * 65 + yy) * 65 + xx) * 1024 + c;
}

// ---- zero the yy=0 plane of the SAT (ws is poisoned, so re-zero each call)
__global__ void zero_row0_kernel(float* __restrict__ sat) {
    int id = blockIdx.x * blockDim.x + threadIdx.x;   // 2*65*1024 = 133120
    if (id >= NB * 65 * 1024) return;
    int n  = id / (65 * 1024);
    int r  = id % (65 * 1024);
    int xx = r / 1024;
    int c  = r % 1024;
    sat[sat_idx(n, 0, xx, c)] = 0.0f;
}

// ---- row-direction prefix sums: sat[n][y+1][x+1][c] = sum_{x'<=x} h[n,c,y,x']
__global__ void sat_rows_kernel(const float* __restrict__ h, float* __restrict__ sat) {
    int bid = blockIdx.x;                 // 512 blocks: (n, y, cChunk)
    int cChunk = bid & 3;
    int y = (bid >> 2) & 63;
    int n = bid >> 8;
    int c = cChunk * 256 + threadIdx.x;
    float acc = 0.0f;
    sat[sat_idx(n, y + 1, 0, c)] = 0.0f;  // xx = 0 column
    const float* row = h + (((size_t)(n * CIN + c) * HH) + y) * WW;
    #pragma unroll 4
    for (int x = 0; x < WW; ++x) {
        acc += row[x];
        sat[sat_idx(n, y + 1, x + 1, c)] = acc;
    }
}

// ---- column-direction prefix sums over the row sums
__global__ void sat_cols_kernel(float* __restrict__ sat) {
    int id = blockIdx.x * blockDim.x + threadIdx.x;   // 2*64*1024 = 131072
    if (id >= NB * 64 * 1024) return;
    int n  = id / (64 * 1024);
    int r  = id % (64 * 1024);
    int xx = 1 + r / 1024;
    int c  = r % 1024;
    float acc = 0.0f;
    for (int yy = 1; yy <= HH; ++yy) {
        int idx = sat_idx(n, yy, xx, c);
        acc += sat[idx];
        sat[idx] = acc;
    }
}

// ---- build K-major weight matrix Bmat[o][k], k = bin*1024 + c, scaled by 1/49
__global__ void prep_w_kernel(const float* __restrict__ Wc, const float* __restrict__ Wl,
                              __bf16* __restrict__ bm) {
    int id = blockIdx.x * blockDim.x + threadIdx.x;   // 112*50176 = 5619712
    if (id >= NPAD * KDIM) return;
    int o = id / KDIM;
    int k = id % KDIM;
    int bin = k >> 10;
    int c   = k & 1023;
    float w = 0.0f;
    if (o < 21)       w = Wc[(size_t)(o * NBIN + bin) * CIN + c];
    else if (o < 105) w = Wl[(size_t)((o - 21) * NBIN + bin) * CIN + c];
    bm[(size_t)o * KDIM + k] = (__bf16)(w * (1.0f / 49.0f));
}

// ---- bias: mean over bins of the per-channel bias (all 49 bins have area>0 here)
__global__ void prep_bias_kernel(const float* __restrict__ bc, const float* __restrict__ bl,
                                 float* __restrict__ bias) {
    int o = threadIdx.x;
    if (o >= NPAD) return;
    float s = 0.0f;
    for (int bin = 0; bin < NBIN; ++bin) {
        if (o < 21)       s += bc[o * NBIN + bin];
        else if (o < 105) s += bl[(o - 21) * NBIN + bin];
    }
    bias[o] = s * (1.0f / 49.0f);
}

// ---- PSRoI bin averages of raw features via SAT: P[r][bin*1024+c] (bf16)
__global__ void pool_kernel(const float* __restrict__ sat,
                            const float* __restrict__ rois,
                            const int*   __restrict__ ridx,
                            __bf16* __restrict__ P) {
    int r   = blockIdx.x;          // one RoI per block
    int tid = threadIdx.x;         // 256 threads
    int b = ridx[r];
    // round-half-away-from-zero (coords non-negative): floor(v*scale + 0.5)
    int x1 = (int)floorf(rois[r * 4 + 0] * SCALE_F + 0.5f);
    int y1 = (int)floorf(rois[r * 4 + 1] * SCALE_F + 0.5f);
    int x2 = (int)floorf(rois[r * 4 + 2] * SCALE_F + 0.5f);
    int y2 = (int)floorf(rois[r * 4 + 3] * SCALE_F + 0.5f);
    float bw = (float)max(x2 - x1, 1) / (float)BINSK;
    float bh = (float)max(y2 - y1, 1) / (float)BINSK;
    __bf16* Pr = P + (size_t)r * KDIM;
    for (int bin = 0; bin < NBIN; ++bin) {
        int i = bin / BINSK, j = bin % BINSK;
        int hs = min(max((int)floorf((float)i * bh) + y1, 0), HH);
        int he = min(max((int)ceilf(((float)i + 1.0f) * bh) + y1, 0), HH);
        int ws_ = min(max((int)floorf((float)j * bw) + x1, 0), WW);
        int we = min(max((int)ceilf(((float)j + 1.0f) * bw) + x1, 0), WW);
        int area = (he - hs) * (we - ws_);
        float inv = (area > 0) ? (1.0f / (float)area) : 0.0f;
        for (int c = tid; c < CIN; c += 256) {
            float s = sat[sat_idx(b, he, we, c)] - sat[sat_idx(b, hs, we, c)]
                    - sat[sat_idx(b, he, ws_, c)] + sat[sat_idx(b, hs, ws_, c)];
            Pr[bin * 1024 + c] = (__bf16)(s * inv);
        }
    }
}

// ---- GEMM: D[1024,112] = P[1024,50176] * Bmat^T, K split 16-way into partials
__global__ void gemm_kernel(const __bf16* __restrict__ P,
                            const __bf16* __restrict__ Bm,
                            float* __restrict__ part) {
    unsigned wid   = (blockIdx.x * blockDim.x + threadIdx.x) >> 5;  // 0..1023
    unsigned lane  = threadIdx.x & 31u;
    unsigned strip = wid & 63u;        // M-tile (16 rows)
    unsigned kseg  = wid >> 6;         // K segment (0..15)
    unsigned m0    = strip * 16u;
    unsigned k0    = kseg * (unsigned)KSEG;
    unsigned l15   = lane & 15u;
    unsigned half  = lane >> 4;

    v8f acc[7];
    #pragma unroll
    for (int t = 0; t < 7; ++t) acc[t] = (v8f){0.f,0.f,0.f,0.f,0.f,0.f,0.f,0.f};

    // A (16x32 bf16): lane holds row m0+l15; K(e) = 16*half + e (contiguous)
    const __bf16* aBase = P + (size_t)(m0 + l15) * KDIM + half * 16u;
    // B (32x16 bf16): lane holds col l15; K(e) = 16*half + e (contiguous)
    const __bf16* bBase = Bm + (size_t)l15 * KDIM + half * 16u;

    for (unsigned kk = k0; kk < k0 + KSEG; kk += 32u) {
        v8bf alo = *(const v8bf*)(aBase + kk);        // K = 16*half + 0..7
        v8bf ahi = *(const v8bf*)(aBase + kk + 8u);   // K = 16*half + 8..15
        v16bf a;
        #pragma unroll
        for (int e = 0; e < 8; ++e) { a[e] = alo[e]; a[e + 8] = ahi[e]; }
        #pragma unroll
        for (int t = 0; t < 7; ++t) {
            v16bf bfr = *(const v16bf*)(bBase + (size_t)t * 16u * KDIM + kk);
            acc[t] = __builtin_amdgcn_wmma_f32_16x16x32_bf16(
                false, a, false, bfr, (short)0, acc[t], false, false);
        }
    }

    // D layout: VGPR v -> m = m0 + v + 8*half, n = t*16 + l15
    #pragma unroll
    for (int t = 0; t < 7; ++t) {
        #pragma unroll
        for (int v = 0; v < 8; ++v) {
            unsigned m = m0 + (unsigned)v + half * 8u;
            unsigned n = (unsigned)t * 16u + l15;
            part[((size_t)kseg * RR + m) * NPAD + n] = acc[t][v];
        }
    }
}

// ---- deterministic K-split reduction + bias + scatter into output layout
__global__ void reduce_kernel(const float* __restrict__ part,
                              const float* __restrict__ bias,
                              float* __restrict__ out) {
    int id = blockIdx.x * blockDim.x + threadIdx.x;   // 1024*112
    if (id >= RR * NPAD) return;
    int r = id / NPAD;
    int o = id % NPAD;
    if (o >= NOUT) return;
    float s = bias[o];
    #pragma unroll
    for (int ks = 0; ks < KSPLIT; ++ks)
        s += part[((size_t)ks * RR + r) * NPAD + o];
    if (o < 21) out[(size_t)r * 21 + o] = s;                       // roi_score
    else        out[(size_t)RR * 21 + (size_t)r * 84 + (o - 21)] = s;  // roi_locs
}

extern "C" void kernel_launch(void* const* d_in, const int* in_sizes, int n_in,
                              void* d_out, int out_size, void* d_ws, size_t ws_size,
                              hipStream_t stream) {
    const float* h    = (const float*)d_in[0];
    const float* rois = (const float*)d_in[1];
    const int*   ridx = (const int*)  d_in[2];
    const float* Wc   = (const float*)d_in[3];
    const float* bc   = (const float*)d_in[4];
    const float* Wl   = (const float*)d_in[5];
    const float* bl   = (const float*)d_in[6];
    float* out = (float*)d_out;

    // workspace layout (256B-aligned offsets)
    char* ws = (char*)d_ws;
    const size_t SAT_BYTES  = (size_t)NB * 65 * 65 * 1024 * 4;   // 34,611,200
    const size_t P_BYTES    = (size_t)RR * KDIM * 2;             // 102,760,448
    const size_t B_BYTES    = (size_t)NPAD * KDIM * 2;           // 11,239,424
    const size_t PART_BYTES = (size_t)KSPLIT * RR * NPAD * 4;    // 7,340,032
    float*  sat  = (float*) (ws);
    __bf16* Pmat = (__bf16*)(ws + SAT_BYTES);
    __bf16* Bmat = (__bf16*)(ws + SAT_BYTES + P_BYTES);
    float*  part = (float*) (ws + SAT_BYTES + P_BYTES + B_BYTES);
    float*  bias = (float*) (ws + SAT_BYTES + P_BYTES + B_BYTES + PART_BYTES);

    zero_row0_kernel<<<520, 256, 0, stream>>>(sat);
    sat_rows_kernel <<<512, 256, 0, stream>>>(h, sat);
    sat_cols_kernel <<<512, 256, 0, stream>>>(sat);
    prep_w_kernel   <<<(NPAD * KDIM) / 256, 256, 0, stream>>>(Wc, Wl, Bmat);
    prep_bias_kernel<<<1, 128, 0, stream>>>(bc, bl, bias);
    pool_kernel     <<<RR, 256, 0, stream>>>(sat, rois, ridx, Pmat);
    gemm_kernel     <<<128, 256, 0, stream>>>(Pmat, Bmat, part);
    reduce_kernel   <<<(RR * NPAD) / 256, 256, 0, stream>>>(part, bias, out);
}
